// DCRNNCell_59167469470014
// MI455X (gfx1250) — matrix operand order repaired and checked by reading, not compiled, not run on hardware
//
#include <hip/hip_runtime.h>
#include <hip/hip_bf16.h>
#include <math.h>

// ---------------- problem constants ----------------
constexpr int Bb   = 32;
constexpr int Nn   = 2048;
constexpr int DESCn= 16;
constexpr int Hh   = 64;
constexpr int Pp   = DESCn + Hh;   // 80
constexpr int KMATn= 5;            // identity + 2 powers * 2 supports
constexpr int Cc   = Bb * Pp;      // 2560 feature rows (transposed layout)
constexpr int KTOT = Pp * KMATn;   // 400 reduction dim of theta GEMM
constexpr size_t XSZ = (size_t)Cc * Nn;        // elements per diffused-feature buffer

typedef __attribute__((ext_vector_type(16))) __bf16 v16bf;
typedef __attribute__((ext_vector_type(8)))  __bf16 v8bf;
typedef __attribute__((ext_vector_type(8)))  float  v8f;

union fragU { v16bf v; v8bf h[2]; };

// LDS byte offset of a shared-memory element (flat addr low 32 bits = LDS offset)
#define LDS_OFF(p) ((unsigned)(uintptr_t)(p))

// async global -> LDS, 16 bytes per lane, GVS mode (SGPR64 base + VGPR byte offset)
#define ASYNC_LDS_B128(ldsoff, gbase, goff)                                  \
  asm volatile("global_load_async_to_lds_b128 %0, %1, %2 offset:0"           \
               :: "v"(ldsoff), "v"(goff), "s"(gbase) : "memory")

#define WAIT_ASYNC0() asm volatile("s_wait_asynccnt 0x0" ::: "memory")
// <=3 outstanding: ASYNCcnt completes in order, so the previous stage's 3 loads are done
#define WAIT_ASYNC3() asm volatile("s_wait_asynccnt 0x3" ::: "memory")

// ---------------- support construction ----------------
__global__ void rowcol_inv_kernel(const float* __restrict__ adj,
                                  float* __restrict__ rinv,
                                  float* __restrict__ cinv) {
  __shared__ float red[256];
  const int n = blockIdx.x;
  const int t = threadIdx.x;
  float rs = 0.f, cs = 0.f;
  for (int m = t; m < Nn; m += 256) {
    rs += adj[(size_t)n * Nn + m];   // row n
    cs += adj[(size_t)m * Nn + n];   // col n
  }
  red[t] = rs; __syncthreads();
  for (int s = 128; s > 0; s >>= 1) { if (t < s) red[t] += red[t + s]; __syncthreads(); }
  if (t == 0) rinv[n] = (red[0] > 0.f) ? 1.f / red[0] : 0.f;
  __syncthreads();
  red[t] = cs; __syncthreads();
  for (int s = 128; s > 0; s >>= 1) { if (t < s) red[t] += red[t + s]; __syncthreads(); }
  if (t == 0) cinv[n] = (red[0] > 0.f) ? 1.f / red[0] : 0.f;
}

// S1 = transpose(rowdiag^-1 * A) ; S2 = transpose(coldiag^-1 * A^T), stored bf16
__global__ void make_supports_kernel(const float* __restrict__ adj,
                                     const float* __restrict__ rinv,
                                     const float* __restrict__ cinv,
                                     __bf16* __restrict__ S1,
                                     __bf16* __restrict__ S2) {
  const size_t idx = (size_t)blockIdx.x * 256 + threadIdx.x;
  const int m = (int)(idx >> 11);
  const int n = (int)(idx & (Nn - 1));
  S1[idx] = (__bf16)(adj[(size_t)n * Nn + m] * rinv[n]);
  S2[idx] = (__bf16)(adj[(size_t)m * Nn + n] * cinv[n]);
}

// Feature-major activations: Xt[c][n], c = b*P + p
__global__ void build_x_kernel(const float* __restrict__ inputs,
                               const float* __restrict__ h,
                               __bf16* __restrict__ X0) {
  const size_t idx = (size_t)blockIdx.x * 256 + threadIdx.x;
  const int n = (int)(idx & (Nn - 1));
  const int c = (int)(idx >> 11);
  const int b = c / Pp;
  const int p = c % Pp;
  float v;
  if (p < DESCn) v = inputs[(size_t)b * Nn * DESCn + (size_t)n * DESCn + p];
  else           v = h[(size_t)b * Nn * Hh + (size_t)n * Hh + (p - DESCn)];
  X0[idx] = (__bf16)v;
}

__global__ void hadamard_kernel(const float* __restrict__ a,
                                const float* __restrict__ b,
                                float* __restrict__ c) {
  const size_t idx = (size_t)blockIdx.x * 256 + threadIdx.x;
  c[idx] = a[idx] * b[idx];
}

// ---------------- diffusion GEMM (transposed):  Yt[c][m] = sum_n Xt[c][n] * S[m][n]
// A = Xt tile (64 c-rows x 32 k), B = S rows (128 m-cols x 32 k) -- both contiguous-K.
// Block = 8 waves, tile 64(c) x 128(m); wave tile 32x32 = 2x2 wmma accumulators.
// Double-buffered async global->LDS staging pipelined against WMMA.
__global__ __launch_bounds__(256) void gemm_diffuse_kernel(const __bf16* __restrict__ S,
                                                           const __bf16* __restrict__ X,
                                                           __bf16* __restrict__ Y) {
  __shared__ __bf16 sA[2][64 * 32];     // [buf][c-row][k]
  __shared__ __bf16 sS[2][128 * 32];    // [buf][m-col][k]
  const int tid  = threadIdx.x;
  const int lane = tid & 31;
  const int w    = tid >> 5;        // 0..7
  const int wm   = w >> 2;          // 0..1  (c-dim)
  const int wn   = w & 3;           // 0..3  (m-dim)
  const int l16  = lane & 15;
  const bool hi  = lane >= 16;
  const int blockM = blockIdx.y * 64;    // c rows
  const int blockN = blockIdx.x * 128;   // m cols

  // async staging addresses (constant per thread, k-offset added per stage)
  const int arow  = tid >> 2, apart = tid & 3;              // A: 64 rows x 4x16B
  const int aoff  = arow * 32 + apart * 8;
  const unsigned aLds[2] = { LDS_OFF(&sA[0][aoff]), LDS_OFF(&sA[1][aoff]) };
  const unsigned aG0  = (unsigned)(((size_t)(blockM + arow) * Nn + apart * 8) * 2);
  const int srow  = tid >> 1;                               // S: 128 rows x 2 lanes x 2x16B
  const int soff  = srow * 32 + (tid & 1) * 16;
  const unsigned sLds[2] = { LDS_OFF(&sS[0][soff]), LDS_OFF(&sS[1][soff]) };
  const unsigned sG0   = (unsigned)(((size_t)(blockN + srow) * Nn + (tid & 1) * 16) * 2);

  const v8f vzero = {0.f,0.f,0.f,0.f,0.f,0.f,0.f,0.f};
  v8f acc[2][2];
  #pragma unroll
  for (int i = 0; i < 2; ++i)
    #pragma unroll
    for (int j = 0; j < 2; ++j) acc[i][j] = vzero;

  constexpr int NSTAGE = Nn / 32;   // 64

  // prologue: stage 0 into buffer 0
  ASYNC_LDS_B128(aLds[0],      X, aG0);
  ASYNC_LDS_B128(sLds[0],      S, sG0);
  ASYNC_LDS_B128(sLds[0] + 16, S, sG0 + 16);

  for (int kt = 0; kt < NSTAGE; ++kt) {
    const int cur = kt & 1;
    const int nxt = cur ^ 1;
    if (kt + 1 < NSTAGE) {      // prefetch next stage into the other buffer
      const unsigned kb = (unsigned)((kt + 1) * 64);   // 32 bf16 = 64 bytes per K-step
      ASYNC_LDS_B128(aLds[nxt],      X, aG0 + kb);
      ASYNC_LDS_B128(sLds[nxt],      S, sG0 + kb);
      ASYNC_LDS_B128(sLds[nxt] + 16, S, sG0 + kb + 16);
      WAIT_ASYNC3();            // current stage landed; next stage still in flight
    } else {
      WAIT_ASYNC0();
    }
    __syncthreads();

    const __bf16* pA = &sA[cur][0];
    const __bf16* pS = &sS[cur][0];
    fragU afrag[2], bfrag[2];
    #pragma unroll
    for (int i = 0; i < 2; ++i) {
      // ISA 16-bit A layout: M=lane&15; K chunks [(hi?8:0)..+8) and +16
      const int m = wm * 32 + i * 16 + l16;
      const int base = m * 32 + (hi ? 8 : 0);
      afrag[i].h[0] = *(const v8bf*)(pA + base);
      afrag[i].h[1] = *(const v8bf*)(pA + base + 16);
    }
    #pragma unroll
    for (int j = 0; j < 2; ++j) {
      // ISA 16-bit B layout: N=lane&15; K chunks [(hi?8:0)..+8) and +16
      const int nn = wn * 32 + j * 16 + l16;
      const int base = nn * 32 + (hi ? 8 : 0);
      bfrag[j].h[0] = *(const v8bf*)(pS + base);
      bfrag[j].h[1] = *(const v8bf*)(pS + base + 16);
    }
    #pragma unroll
    for (int i = 0; i < 2; ++i)
      #pragma unroll
      for (int j = 0; j < 2; ++j)
        acc[i][j] = __builtin_amdgcn_wmma_f32_16x16x32_bf16(
            false, afrag[i].v, false, bfrag[j].v, (short)0, acc[i][j], false, false);
    __syncthreads();   // protects buffer reuse two stages later
  }

  // D layout: VGPR v -> row(c) = v (+8 for hi lanes), col(m) = lane&15
  #pragma unroll
  for (int i = 0; i < 2; ++i)
    #pragma unroll
    for (int j = 0; j < 2; ++j)
      #pragma unroll
      for (int v = 0; v < 8; ++v) {
        const int row = blockM + wm * 32 + i * 16 + (hi ? v + 8 : v);
        const int col = blockN + wn * 32 + j * 16 + l16;
        Y[(size_t)row * Nn + col] = (__bf16)acc[i][j][v];
      }
}

// ---------------- theta GEMM (r & u fused):  [B*N,400] x [400,64], sigmoid ----------------
__global__ __launch_bounds__(256) void gemm_theta_ru_kernel(
    const __bf16* __restrict__ Xbase,                 // 5 buffers of XSZ, layout [c][n]
    const float* __restrict__ theta_r, const float* __restrict__ theta_u,
    const float* __restrict__ bias_r,  const float* __restrict__ bias_u,
    float* __restrict__ rbuf, float* __restrict__ ubuf) {
  const int tid = threadIdx.x, lane = tid & 31, w = tid >> 5;
  const int l16 = lane & 15;
  const bool hi = lane >= 16;
  const int rowbase = (blockIdx.x * 8 + w) * 16;
  const int r = rowbase + l16;
  const int b = r >> 11, n = r & (Nn - 1);

  const v8f vzero = {0.f,0.f,0.f,0.f,0.f,0.f,0.f,0.f};
  v8f acc[2][4];
  #pragma unroll
  for (int t = 0; t < 2; ++t)
    #pragma unroll
    for (int j = 0; j < 4; ++j) acc[t][j] = vzero;

  for (int kt = 0; kt < 13; ++kt) {    // K padded 400 -> 416
    const int k0 = kt * 32;
    v16bf a;
    #pragma unroll
    for (int e = 0; e < 16; ++e) {
      const int q = k0 + e + (hi ? 16 : 0);
      __bf16 v = (__bf16)0.f;
      if (q < KTOT) {
        const int p = q / KMATn, kk = q % KMATn;
        v = Xbase[(size_t)kk * XSZ + (size_t)(b * Pp + p) * Nn + n];
      }
      a[e] = v;
    }
    #pragma unroll
    for (int j = 0; j < 4; ++j) {
      const int h = j * 16 + l16;
      v16bf br, bu;
      #pragma unroll
      for (int e = 0; e < 16; ++e) {
        const int q = k0 + e + (hi ? 16 : 0);
        float fr = 0.f, fu = 0.f;
        if (q < KTOT) { fr = theta_r[q * Hh + h]; fu = theta_u[q * Hh + h]; }
        br[e] = (__bf16)fr;
        bu[e] = (__bf16)fu;
      }
      acc[0][j] = __builtin_amdgcn_wmma_f32_16x16x32_bf16(
          false, a, false, br, (short)0, acc[0][j], false, false);
      acc[1][j] = __builtin_amdgcn_wmma_f32_16x16x32_bf16(
          false, a, false, bu, (short)0, acc[1][j], false, false);
    }
  }

  #pragma unroll
  for (int j = 0; j < 4; ++j) {
    const int h = j * 16 + l16;
    const float brv = bias_r[h], buv = bias_u[h];
    #pragma unroll
    for (int v = 0; v < 8; ++v) {
      const int row = rowbase + (hi ? v + 8 : v);
      const float xr = acc[0][j][v] + brv;
      const float xu = acc[1][j][v] + buv;
      rbuf[(size_t)row * Hh + h] = 1.f / (1.f + expf(-xr));
      ubuf[(size_t)row * Hh + h] = 1.f / (1.f + expf(-xu));
    }
  }
}

// ---------------- theta GEMM (c) + fused GRU output:  out = u*s + (1-u)*tanh(.) ----------------
__global__ __launch_bounds__(256) void gemm_theta_c_kernel(
    const __bf16* __restrict__ Xbase,
    const float* __restrict__ theta_c, const float* __restrict__ bias_c,
    const float* __restrict__ ubuf,    const float* __restrict__ state,
    float* __restrict__ out) {
  const int tid = threadIdx.x, lane = tid & 31, w = tid >> 5;
  const int l16 = lane & 15;
  const bool hi = lane >= 16;
  const int rowbase = (blockIdx.x * 8 + w) * 16;
  const int r = rowbase + l16;
  const int b = r >> 11, n = r & (Nn - 1);

  const v8f vzero = {0.f,0.f,0.f,0.f,0.f,0.f,0.f,0.f};
  v8f acc[4];
  #pragma unroll
  for (int j = 0; j < 4; ++j) acc[j] = vzero;

  for (int kt = 0; kt < 13; ++kt) {
    const int k0 = kt * 32;
    v16bf a;
    #pragma unroll
    for (int e = 0; e < 16; ++e) {
      const int q = k0 + e + (hi ? 16 : 0);
      __bf16 v = (__bf16)0.f;
      if (q < KTOT) {
        const int p = q / KMATn, kk = q % KMATn;
        v = Xbase[(size_t)kk * XSZ + (size_t)(b * Pp + p) * Nn + n];
      }
      a[e] = v;
    }
    #pragma unroll
    for (int j = 0; j < 4; ++j) {
      const int h = j * 16 + l16;
      v16bf bc;
      #pragma unroll
      for (int e = 0; e < 16; ++e) {
        const int q = k0 + e + (hi ? 16 : 0);
        bc[e] = (__bf16)((q < KTOT) ? theta_c[q * Hh + h] : 0.f);
      }
      acc[j] = __builtin_amdgcn_wmma_f32_16x16x32_bf16(
          false, a, false, bc, (short)0, acc[j], false, false);
    }
  }

  #pragma unroll
  for (int j = 0; j < 4; ++j) {
    const int h = j * 16 + l16;
    const float bcv = bias_c[h];
    #pragma unroll
    for (int v = 0; v < 8; ++v) {
      const int row = rowbase + (hi ? v + 8 : v);
      const size_t o = (size_t)row * Hh + h;
      const float c = tanhf(acc[j][v] + bcv);
      const float u = ubuf[o];
      out[o] = u * state[o] + (1.f - u) * c;
    }
  }
}

// ---------------- launcher ----------------
extern "C" void kernel_launch(void* const* d_in, const int* in_sizes, int n_in,
                              void* d_out, int out_size, void* d_ws, size_t ws_size,
                              hipStream_t stream) {
  (void)in_sizes; (void)n_in; (void)out_size; (void)ws_size;
  const float* inputs  = (const float*)d_in[0];
  const float* state   = (const float*)d_in[1];
  const float* adj     = (const float*)d_in[2];
  const float* theta_r = (const float*)d_in[3];
  const float* theta_u = (const float*)d_in[4];
  const float* theta_c = (const float*)d_in[5];
  const float* bias_r  = (const float*)d_in[6];
  const float* bias_u  = (const float*)d_in[7];
  const float* bias_c  = (const float*)d_in[8];
  float* out = (float*)d_out;

  char* ws = (char*)d_ws;
  size_t off = 0;
  float*  rinv = (float*)(ws + off);  off += (size_t)Nn * 4;
  float*  cinv = (float*)(ws + off);  off += (size_t)Nn * 4;
  __bf16* S1   = (__bf16*)(ws + off); off += (size_t)Nn * Nn * 2;
  __bf16* S2   = (__bf16*)(ws + off); off += (size_t)Nn * Nn * 2;
  __bf16* X    = (__bf16*)(ws + off); off += 5 * XSZ * 2;   // X+k*XSZ = k-th diffused buffer
  float*  rbuf = (float*)(ws + off);  off += (size_t)Bb * Nn * Hh * 4;
  float*  ubuf = (float*)(ws + off);  off += (size_t)Bb * Nn * Hh * 4;
  float*  hc   = (float*)(ws + off);  off += (size_t)Bb * Nn * Hh * 4;

  const dim3 blk(256);
  const dim3 gdiff(Nn / 128, Cc / 64);                 // 16 x 40
  const int  gsup  = (Nn * Nn) / 256;                  // 16384
  const int  gx    = (int)(XSZ / 256);                 // 20480
  const int  ghad  = (Bb * Nn * Hh) / 256;             // 16384
  const int  gtht  = (Bb * Nn) / 128;                  // 512 (8 waves * 16 rows)

  // supports
  rowcol_inv_kernel<<<Nn, blk, 0, stream>>>(adj, rinv, cinv);
  make_supports_kernel<<<gsup, blk, 0, stream>>>(adj, rinv, cinv, S1, S2);

  // gconv for r/u (h = state)
  build_x_kernel<<<gx, blk, 0, stream>>>(inputs, state, X);
  gemm_diffuse_kernel<<<gdiff, blk, 0, stream>>>(S1, X,           X + XSZ);      // S1 x
  gemm_diffuse_kernel<<<gdiff, blk, 0, stream>>>(S1, X + XSZ,     X + 2 * XSZ);  // S1^2 x
  gemm_diffuse_kernel<<<gdiff, blk, 0, stream>>>(S2, X,           X + 3 * XSZ);  // S2 x
  gemm_diffuse_kernel<<<gdiff, blk, 0, stream>>>(S2, X + 3 * XSZ, X + 4 * XSZ);  // S2^2 x
  gemm_theta_ru_kernel<<<gtht, blk, 0, stream>>>(X, theta_r, theta_u, bias_r, bias_u,
                                                 rbuf, ubuf);

  // gconv for c (h = r * state)
  hadamard_kernel<<<ghad, blk, 0, stream>>>(rbuf, state, hc);
  build_x_kernel<<<gx, blk, 0, stream>>>(inputs, hc, X);
  gemm_diffuse_kernel<<<gdiff, blk, 0, stream>>>(S1, X,           X + XSZ);
  gemm_diffuse_kernel<<<gdiff, blk, 0, stream>>>(S1, X + XSZ,     X + 2 * XSZ);
  gemm_diffuse_kernel<<<gdiff, blk, 0, stream>>>(S2, X,           X + 3 * XSZ);
  gemm_diffuse_kernel<<<gdiff, blk, 0, stream>>>(S2, X + 3 * XSZ, X + 4 * XSZ);
  gemm_theta_c_kernel<<<gtht, blk, 0, stream>>>(X, theta_c, bias_c, ubuf, state, out);
}
